// ProteinMPNNEncoder_73701638799792
// MI455X (gfx1250) — matrix operand order repaired and checked by reading, not compiled, not run on hardware
//
#include <hip/hip_runtime.h>
#include <hip/hip_bf16.h>
#include <math.h>

typedef __attribute__((ext_vector_type(16))) _Float16 v16h;
typedef __attribute__((ext_vector_type(8)))  float    v8f;

#define Bsz 2
#define Lsz 2048
#define Ksz 32
#define NBINS 37
#define EDGE_DIM 64
#define NDIST 925              // 5*5*37
#define NREL 65                // rel in [-32,32]

// ---------------------------------------------------------------------------
// Kernel 0: build all_coords (B,L,5,3): 4 input atoms + virtual CB.
// ---------------------------------------------------------------------------
__global__ void build_allcoords(const float* __restrict__ coords,
                                float* __restrict__ allc, int BL) {
  int i = blockIdx.x * blockDim.x + threadIdx.x;
  if (i >= BL) return;
  const float* c = coords + (size_t)i * 12;
  float nx = c[0], ny = c[1], nz = c[2];
  float ax = c[3], ay = c[4], az = c[5];
  float cx = c[6], cy = c[7], cz = c[8];
  // ca_n = normalize(N - CA)
  float ux = nx - ax, uy = ny - ay, uz = nz - az;
  float inv = 1.0f / fmaxf(sqrtf(ux*ux + uy*uy + uz*uz), 1e-12f);
  ux *= inv; uy *= inv; uz *= inv;
  // ca_c = normalize(C - CA)
  float vx = cx - ax, vy = cy - ay, vz = cz - az;
  inv = 1.0f / fmaxf(sqrtf(vx*vx + vy*vy + vz*vz), 1e-12f);
  vx *= inv; vy *= inv; vz *= inv;
  // bisector
  float bx = ux + vx, by = uy + vy, bz = uz + vz;
  inv = 1.0f / fmaxf(sqrtf(bx*bx + by*by + bz*bz), 1e-12f);
  bx *= inv; by *= inv; bz *= inv;
  // perpendicular = normalize(cross(ca_n, ca_c))
  float px = uy*vz - uz*vy, py = uz*vx - ux*vz, pz = ux*vy - uy*vx;
  inv = 1.0f / fmaxf(sqrtf(px*px + py*py + pz*pz), 1e-12f);
  px *= inv; py *= inv; pz *= inv;
  // cb_dir = normalize(-bisector + 0.5*perp); virtual_cb = ca + 1.54*cb_dir
  float dx = -bx + 0.5f*px, dy = -by + 0.5f*py, dz = -bz + 0.5f*pz;
  inv = 1.0f / fmaxf(sqrtf(dx*dx + dy*dy + dz*dz), 1e-12f);
  float ex = ax + 1.54f*dx*inv, ey = ay + 1.54f*dy*inv, ez = az + 1.54f*dz*inv;

  float* o = allc + (size_t)i * 15;
#pragma unroll
  for (int t = 0; t < 12; ++t) o[t] = c[t];
  o[12] = ex; o[13] = ey; o[14] = ez;
}

// ---------------------------------------------------------------------------
// Kernel 1: PE table via WMMA.  PE_table[r][h] = pe(r-32) @ W_pe + b_edge,
// r = 0..64.  M=65 padded to 80 -> 5 row tiles x 4 col tiles = 20 waves.
// Each wave: two chained v_wmma_f32_16x16x32_f16 (K = 64).
// ---------------------------------------------------------------------------
__device__ __forceinline__ float pe_val(float rel, int p) {
  // div_term for feature p: exp(-(p&~1) * ln(10000)/64); even->sin, odd->cos
  float ang = rel * __expf(-(float)(p & ~1) * (9.210340371976184f / 64.0f));
  return (p & 1) ? __cosf(ang) : __sinf(ang);
}

__global__ void build_pe_table(const float* __restrict__ W_edge,
                               const float* __restrict__ b_edge,
                               float* __restrict__ pe_tab) {
  const int lane = threadIdx.x;           // 32 threads = 1 wave per block
  const int tile = blockIdx.x;            // 0..19
  const int mt = tile >> 2, nt = tile & 3;
  const int g  = lane >> 4;               // half-wave group
  const int nl = lane & 15;

  const int M_a = mt * 16 + nl;           // A-matrix row held by this lane
  float rel = (float)((M_a > 64 ? 64 : M_a) - 32);   // clamp pad rows

  v8f acc = {};
#pragma unroll
  for (int kb = 0; kb < 2; ++kb) {
    v16h a, b;
#pragma unroll
    for (int e = 0; e < 16; ++e) {
      // A 16x32 f16 layout (ISA 7.12.2): VGPR v=e/2, half h=e%2
      int v = e >> 1, h = e & 1;
      int Ka = (v < 4) ? (8*g + 2*v + h) : (16 + 8*g + 2*(v - 4) + h);
      a[e] = (_Float16)pe_val(rel, kb*32 + Ka);
      // B 32x16 f16 layout: K = 8*g + e (e<8), 16 + 8*g + (e-8) (e>=8), N = lane&15
      int Kb = kb*32 + ((e < 8) ? (8*g + e) : (16 + 8*g + (e - 8)));
      b[e] = (_Float16)W_edge[(size_t)(NDIST + Kb) * EDGE_DIM + nt*16 + nl];
    }
    acc = __builtin_amdgcn_wmma_f32_16x16x32_f16(
        /*neg_a=*/false, a, /*neg_b=*/false, b,
        /*c_mod=*/(short)0, acc, /*reuse_a=*/false, /*reuse_b=*/false);
  }

  // C/D layout: element i -> M = 8*g + i, N = lane&15
  float bias = b_edge[nt*16 + nl];
#pragma unroll
  for (int i = 0; i < 8; ++i) {
    int M = mt*16 + 8*g + i;
    if (M < NREL) pe_tab[M * EDGE_DIM + nt*16 + nl] = acc[i] + bias;
  }
}

// ---------------------------------------------------------------------------
// Kernel 2: main edge kernel.  One wave per (b,l,k).  Lanes 0..24 compute one
// atom-pair distance bin each; rows broadcast via v_readlane; all lanes
// accumulate 2 channels (float2) -> fully coalesced 256B gathers & stores.
// ---------------------------------------------------------------------------
__global__ void edge_main(const float* __restrict__ allc,
                          const int*   __restrict__ nb,
                          const float* __restrict__ W_edge,
                          const float* __restrict__ pe_tab,
                          float* __restrict__ out) {
  const int lane = threadIdx.x & 31;
  const int wid  = blockIdx.x * (blockDim.x >> 5) + (threadIdx.x >> 5);
  const int k  = wid % Ksz;
  const int bl = wid / Ksz;                // b*L + l
  const int l  = bl % Lsz;

  const int j  = nb[(size_t)bl * Ksz + k]; // neighbor residue (wave-uniform)
  const int bj = (bl - l) + j;             // b*L + j

  unsigned row = 0;
  if (lane < 25) {
    const int ai = lane / 5, aj = lane % 5;
    const float* pc = allc + (size_t)bl * 15 + ai * 3;
    const float* pn = allc + (size_t)bj * 15 + aj * 3;
    float dx = pc[0] - pn[0], dy = pc[1] - pn[1], dz = pc[2] - pn[2];
    float d  = sqrtf(dx*dx + dy*dy + dz*dz);
    // searchsorted(linspace(2,20,37), d, 'left') clipped to [0,36]
    int bin = (int)ceilf((d - 2.0f) * 2.0f);
    bin = bin < 0 ? 0 : (bin > NBINS - 1 ? NBINS - 1 : bin);
    row = (unsigned)(lane * NBINS + bin);  // lane == ai*5+aj
  }

  int rel = j - l;
  rel = rel < -32 ? -32 : (rel > 32 ? 32 : rel);

  const float2* pe2 = (const float2*)pe_tab;
  float2 acc = pe2[(size_t)(rel + 32) * 32 + lane];   // pe@W_pe + b_edge

  const float2* W2 = (const float2*)W_edge;           // rows 0..924 = W_dist
#pragma unroll
  for (int p = 0; p < 25; ++p) {
    unsigned r = (unsigned)__builtin_amdgcn_readlane((int)row, p); // scalar bcast
    float2 w = W2[(size_t)r * 32 + lane];
    acc.x += w.x; acc.y += w.y;
  }

  ((float2*)out)[(size_t)wid * 32 + lane] = acc;
}

// ---------------------------------------------------------------------------
extern "C" void kernel_launch(void* const* d_in, const int* in_sizes, int n_in,
                              void* d_out, int out_size, void* d_ws, size_t ws_size,
                              hipStream_t stream) {
  const float* coords = (const float*)d_in[0];   // (B,L,4,3) f32
  const int*   nbi    = (const int*)  d_in[1];   // (B,L,K)   i32
  const float* W      = (const float*)d_in[2];   // (989,64)  f32
  const float* be     = (const float*)d_in[3];   // (64,)     f32
  float* out = (float*)d_out;

  float* allc   = (float*)d_ws;                  // B*L*15 floats = 245,760 B
  float* pe_tab = allc + (size_t)Bsz * Lsz * 15; // 65*64 floats  =  16,640 B

  const int BL = Bsz * Lsz;
  build_allcoords<<<(BL + 255) / 256, 256, 0, stream>>>(coords, allc, BL);
  build_pe_table<<<20, 32, 0, stream>>>(W, be, pe_tab);

  const int waves = Bsz * Lsz * Ksz;             // 131072, one wave per row
  edge_main<<<waves / 8, 256, 0, stream>>>(allc, nbi, W, pe_tab, out);
}
